// GraphNet_25993142075822
// MI455X (gfx1250) — matrix operand order repaired and checked by reading, not compiled, not run on hardware
//
#include <hip/hip_runtime.h>

typedef __attribute__((ext_vector_type(16))) __bf16 v16bf;
typedef __attribute__((ext_vector_type(2)))  __bf16 bf16x2;
typedef __attribute__((ext_vector_type(8)))  float  v8f;
typedef __attribute__((ext_vector_type(2)))  float  v2f;

#define A_DIM 16
#define M_DIM 64
#define H_DIM 128
#define O_DIM 64
#define WAVES_PER_WG 8
#define NBLOCKS 512
#define WAVE_F32 2048          // 16x128 f32 tile per wave

// LDS u32 offsets for staged bf16 weights (WMMA-B fragment layout)
#define WG1_OFF 0              // W_gc1: 2 kc * 8 nt  -> 4096 u32
#define WF1_OFF 4096           // W_fc1: 2 kc * 8 nt  -> 4096 u32
#define WG2_OFF 8192           // W_gc2: 4 kc * 8 nt  -> 8192 u32
#define WF2_OFF 16384          // W_fc2: 4 kc * 8 nt  -> 8192 u32
#define WF3_OFF 24576          // W_fc3: 4 kc * 4 nt  -> 4096 u32
#define WTOT_U32 28672
// bias block (f32 indices within blds)
#define B_GC1 0
#define B_FC1 128
#define B_GC2 256
#define B_FC2 384
#define B_FC3 512
#define B_TOT 576

union BFrag { unsigned u[8]; v16bf v; };

// One packed bf16 pair via the native f32->bf16 convert (v_cvt_pk_bf16_f32 class).
__device__ __forceinline__ unsigned pack2bf(float a, float b) {
    bf16x2 t;
    t[0] = (__bf16)a;
    t[1] = (__bf16)b;
    return __builtin_bit_cast(unsigned, t);
}

__device__ __forceinline__ float relu(float v) {
    return __builtin_amdgcn_fmed3f(v, 0.0f, __builtin_inff());   // one v_med3_f32
}

// Stage a (32*ktiles) x ncols f32 weight matrix into LDS as packed bf16 WMMA-B
// fragments: frag(kc,nt) = kc*ntiles+nt, lane L holds col N = nt*16 + L%16,
// slot v packs (K,K+1) with K = kc*32 + 16*(L/16) + 2v.
__device__ __forceinline__ void stage_weight(unsigned* dst, const float* W,
                                             int ktiles, int ntiles, int ncols, int tid) {
    const int total = ktiles * ntiles * 256;
    for (int i = tid; i < total; i += 256) {
        int v = i & 7, l = (i >> 3) & 31, frag = i >> 8;
        int kc = frag / ntiles, nt = frag - kc * ntiles;
        int K = kc * 32 + 16 * (l >> 4) + 2 * v;
        int N = nt * 16 + (l & 15);
        dst[frag * 256 + l * 8 + v] = pack2bf(W[K * ncols + N], W[(K + 1) * ncols + N]);
    }
}

// Load one 32-byte B fragment with two b128 LDS reads.
__device__ __forceinline__ BFrag load_bfrag(const unsigned* base) {
    BFrag f;
    const uint4* p = (const uint4*)base;
    uint4 a = p[0], b = p[1];
    f.u[0] = a.x; f.u[1] = a.y; f.u[2] = a.z; f.u[3] = a.w;
    f.u[4] = b.x; f.u[5] = b.y; f.u[6] = b.z; f.u[7] = b.w;
    return f;
}

// Build bf16 A fragments for row M = lane%16 from a contiguous f32 row
// (global or LDS). K order used here (contiguous): K = 16*half+2v (matches B staging).
template <int NKC>
__device__ __forceinline__ void build_afrag(BFrag* A, const float* row, int half) {
    #pragma unroll
    for (int kc = 0; kc < NKC; ++kc)
        #pragma unroll
        for (int v = 0; v < 8; ++v) {
            int K = kc * 32 + 16 * half + 2 * v;
            float2 p = *(const float2*)(row + K);
            A[kc].u[v] = pack2bf(p.x, p.y);
        }
}

// Splat a scalar into all 8 accumulator slots (bias pre-loaded as WMMA C input).
__device__ __forceinline__ v8f splat8(float s) {
    v8f r;
    #pragma unroll
    for (int i = 0; i < 8; ++i) r[i] = s;
    return r;
}

// One 16-col adjacency tile: acc = adj^T(16x16) @ h(16xH)[:, col-tile],
// as 4 chained f32 16x16x4 WMMAs reading h from the wave's LDS tile.
__device__ __forceinline__ v8f adj_tile(const v2f* adjA, const float* hb, int col, int half) {
    v8f acc = {};
    #pragma unroll
    for (int c = 0; c < 4; ++c) {
        int k0 = 4 * c + 2 * half;
        v2f Bf;
        Bf[0] = hb[k0 * H_DIM + col];
        Bf[1] = hb[(k0 + 1) * H_DIM + col];
        acc = __builtin_amdgcn_wmma_f32_16x16x4_f32(false, adjA[c], false, Bf, (short)0, acc, false, false);
    }
    return acc;
}

__global__ void graphnet_fused(
    const float* __restrict__ x,    const float* __restrict__ adj,
    const float* __restrict__ Wgc1, const float* __restrict__ bgc1,
    const float* __restrict__ Wfc1, const float* __restrict__ bfc1,
    const float* __restrict__ Wgc2, const float* __restrict__ bgc2,
    const float* __restrict__ Wfc2, const float* __restrict__ bfc2,
    const float* __restrict__ Wfc3, const float* __restrict__ bfc3,
    float* __restrict__ out)
{
    extern __shared__ unsigned char smem_raw[];
    unsigned* wlds = (unsigned*)smem_raw;                 // 28672 u32 staged weights
    float* blds = (float*)(wlds + WTOT_U32);              // 576 f32 staged biases
    float* scratch = blds + B_TOT;                        // 8 waves * 2048 f32

    const int tid  = threadIdx.x;
    const int lane = tid & 31;
    const int wid  = tid >> 5;
    const int half = lane >> 4;
    const int l16  = lane & 15;

    stage_weight(wlds + WG1_OFF, Wgc1, 2, 8, H_DIM, tid);
    stage_weight(wlds + WF1_OFF, Wfc1, 2, 8, H_DIM, tid);
    stage_weight(wlds + WG2_OFF, Wgc2, 4, 8, H_DIM, tid);
    stage_weight(wlds + WF2_OFF, Wfc2, 4, 8, H_DIM, tid);
    stage_weight(wlds + WF3_OFF, Wfc3, 4, 4, O_DIM, tid);
    if (tid < 128) {
        blds[B_GC1 + tid] = bgc1[tid];
        blds[B_FC1 + tid] = bfc1[tid];
        blds[B_GC2 + tid] = bgc2[tid];
        blds[B_FC2 + tid] = bfc2[tid];
        if (tid < 64) blds[B_FC3 + tid] = bfc3[tid];
    }
    __syncthreads();

    float* hbuf = scratch + wid * WAVE_F32;               // 16x128 tile: h -> act, reused in place
    const int gw = blockIdx.x * WAVES_PER_WG + wid;       // 0..4095

    for (int it = 0; it < 2; ++it) {
        const int bt = gw + 4096 * it;
        const float* xbt  = x   + (size_t)bt * A_DIM * M_DIM;
        const float* adjb = adj + (size_t)bt * A_DIM * A_DIM;
        if (it == 0)
            __builtin_prefetch(x + (size_t)(bt + 4096) * A_DIM * M_DIM + lane * 8, 0, 1);

        // ---- A fragments of x (16x64) and adj^T (reused by both adjacency stages)
        BFrag A1[2];
        build_afrag<2>(A1, xbt + l16 * M_DIM, half);
        v2f adjA[4];
        #pragma unroll
        for (int c = 0; c < 4; ++c) {
            int i0 = 4 * c + 2 * half;
            adjA[c][0] = adjb[i0 * A_DIM + l16];          // adj^T[j,i] = adj[i,j]
            adjA[c][1] = adjb[(i0 + 1) * A_DIM + l16];
        }

        // ---- Layer 1, fully fused per column-tile:
        //   h1 tile = x@W_gc1 + b_gc1 ; y3 = (x@W_fc1)[0,col] + b_fc1   (biases ride in C)
        //   -> LDS -> y tile = relu(adj^T@h1 + e0*y3) (4 f32 WMMA) -> LDS (in place)
        #pragma unroll
        for (int nt = 0; nt < 8; ++nt) {
            int col = nt * 16 + l16;
            BFrag Bh0 = load_bfrag(&wlds[WG1_OFF + (0 * 8 + nt) * 256 + lane * 8]);
            BFrag By0 = load_bfrag(&wlds[WF1_OFF + (0 * 8 + nt) * 256 + lane * 8]);
            BFrag Bh1 = load_bfrag(&wlds[WG1_OFF + (1 * 8 + nt) * 256 + lane * 8]);
            BFrag By1 = load_bfrag(&wlds[WF1_OFF + (1 * 8 + nt) * 256 + lane * 8]);
            v8f ah = splat8(blds[B_GC1 + col]);           // bias as initial C
            v8f ay = splat8(blds[B_FC1 + col]);
            ah = __builtin_amdgcn_wmma_f32_16x16x32_bf16(false, A1[0].v, false, Bh0.v, (short)0, ah, false, false);
            ay = __builtin_amdgcn_wmma_f32_16x16x32_bf16(false, A1[0].v, false, By0.v, (short)0, ay, false, false);
            ah = __builtin_amdgcn_wmma_f32_16x16x32_bf16(false, A1[1].v, false, Bh1.v, (short)0, ah, false, false);
            ay = __builtin_amdgcn_wmma_f32_16x16x32_bf16(false, A1[1].v, false, By1.v, (short)0, ay, false, false);
            #pragma unroll
            for (int r = 0; r < 8; ++r)
                hbuf[(r + 8 * half) * H_DIM + col] = ah[r];
            float y3 = ay[0];                             // row 0 lives at r=0, lanes 0-15
            v8f a = adj_tile(adjA, hbuf, col, half);
            if (half == 0) a[0] += y3;
            #pragma unroll
            for (int r = 0; r < 8; ++r)
                hbuf[(r + 8 * half) * H_DIM + col] = relu(a[r]);
        }

        // ---- Layer 2 A fragments from y (K=128 -> 4 frags)
        BFrag A2[4];
        build_afrag<4>(A2, hbuf + l16 * H_DIM, half);

        // ---- Layer 2, fused per column-tile (h2 GEMM + z3 + adjacency + relu)
        #pragma unroll
        for (int nt = 0; nt < 8; ++nt) {
            int col = nt * 16 + l16;
            v8f ah = splat8(blds[B_GC2 + col]);
            v8f az = splat8(blds[B_FC2 + col]);
            #pragma unroll
            for (int kc = 0; kc < 4; ++kc) {
                BFrag Bh = load_bfrag(&wlds[WG2_OFF + (kc * 8 + nt) * 256 + lane * 8]);
                BFrag Bz = load_bfrag(&wlds[WF2_OFF + (kc * 8 + nt) * 256 + lane * 8]);
                ah = __builtin_amdgcn_wmma_f32_16x16x32_bf16(false, A2[kc].v, false, Bh.v, (short)0, ah, false, false);
                az = __builtin_amdgcn_wmma_f32_16x16x32_bf16(false, A2[kc].v, false, Bz.v, (short)0, az, false, false);
            }
            #pragma unroll
            for (int r = 0; r < 8; ++r)
                hbuf[(r + 8 * half) * H_DIM + col] = ah[r];
            float z3 = az[0];
            v8f a = adj_tile(adjA, hbuf, col, half);
            if (half == 0) a[0] += z3;
            #pragma unroll
            for (int r = 0; r < 8; ++r)
                hbuf[(r + 8 * half) * H_DIM + col] = relu(a[r]);
        }

        // ---- v = mean_rows(u @ W_fc3) + b_fc3   (all 16 WMMA rows useful:
        //      z@W = mean_rows(u)@W = mean_rows(u@W))
        BFrag A3[4];
        build_afrag<4>(A3, hbuf + l16 * H_DIM, half);
        #pragma unroll
        for (int nt = 0; nt < 4; ++nt) {
            int col = nt * 16 + l16;
            v8f ap = {};
            #pragma unroll
            for (int kc = 0; kc < 4; ++kc) {
                BFrag Bp = load_bfrag(&wlds[WF3_OFF + (kc * 4 + nt) * 256 + lane * 8]);
                ap = __builtin_amdgcn_wmma_f32_16x16x32_bf16(false, A3[kc].v, false, Bp.v, (short)0, ap, false, false);
            }
            float s = 0.f;
            #pragma unroll
            for (int r = 0; r < 8; ++r) s += ap[r];       // rows r+8*half of this half
            s += __shfl_xor(s, 16, 32);                   // add the other lane-half's rows
            if (half == 0)
                out[(size_t)bt * O_DIM + col] = s * (1.0f / 16.0f) + blds[B_FC3 + col];
        }
    }
}

extern "C" void kernel_launch(void* const* d_in, const int* in_sizes, int n_in,
                              void* d_out, int out_size, void* d_ws, size_t ws_size,
                              hipStream_t stream) {
    (void)in_sizes; (void)n_in; (void)d_ws; (void)ws_size; (void)out_size;
    const float* x    = (const float*)d_in[0];
    const float* adj  = (const float*)d_in[1];
    const float* Wgc1 = (const float*)d_in[2];
    const float* bgc1 = (const float*)d_in[3];
    const float* Wfc1 = (const float*)d_in[4];
    const float* bfc1 = (const float*)d_in[5];
    const float* Wgc2 = (const float*)d_in[6];
    const float* bgc2 = (const float*)d_in[7];
    const float* Wfc2 = (const float*)d_in[8];
    const float* bfc2 = (const float*)d_in[9];
    const float* Wfc3 = (const float*)d_in[10];
    const float* bfc3 = (const float*)d_in[11];
    float* out = (float*)d_out;

    const size_t shmem = (size_t)WTOT_U32 * 4 + (size_t)B_TOT * 4
                       + (size_t)WAVES_PER_WG * WAVE_F32 * 4;   // 182,528 B
    (void)hipFuncSetAttribute((const void*)graphnet_fused,
                              hipFuncAttributeMaxDynamicSharedMemorySize, (int)shmem);
    hipLaunchKernelGGL(graphnet_fused, dim3(NBLOCKS), dim3(WAVES_PER_WG * 32), shmem, stream,
                       x, adj, Wgc1, bgc1, Wfc1, bfc1, Wgc2, bgc2, Wfc2, bfc2, Wfc3, bfc3, out);
}
